// JiTViD_8340826489085
// MI455X (gfx1250) — matrix operand order repaired and checked by reading, not compiled, not run on hardware
//
#include <hip/hip_runtime.h>
#include <hip/hip_bf16.h>
#include <math.h>

typedef __attribute__((ext_vector_type(16))) _Float16 v16h;
typedef __attribute__((ext_vector_type(8)))  float    v8f;
typedef __attribute__((ext_vector_type(4)))  int      v4i;

#define BD   2
#define SEQ  2048
#define CH   768
#define NH   12
#define HD   64
#define M3   2048
#define SROWS (BD*SEQ)
#define EPSF 1e-6f

__device__ __forceinline__ v8f wmma16(v16h a, v16h b, v8f c) {
  // D = A(16x32 f16) * B(32x16 f16) + C(16x16 f32)
  return __builtin_amdgcn_wmma_f32_16x16x32_f16(false, a, false, b, (short)0, c,
                                                false, false);
}

union AF { v16h v; uint4 u[2]; };

// ---------------------------------------------------------------------------
// Async global->LDS staging (CDNA5 GLOBAL_LOAD_ASYNC_TO_LDS_B128, ASYNCcnt).
// Falls back to synchronous copies if the builtins are unavailable.
// ---------------------------------------------------------------------------
#if defined(__has_builtin)
#if __has_builtin(__builtin_amdgcn_global_load_async_to_lds_b128)
#define HAVE_ASYNC_LDS 1
#endif
#endif

typedef __attribute__((address_space(1))) v4i* gptr_v4i;
typedef __attribute__((address_space(3))) v4i* lptr_v4i;

__device__ __forceinline__ void async_ld16(void* ldsDst, const void* gSrc) {
#ifdef HAVE_ASYNC_LDS
  __builtin_amdgcn_global_load_async_to_lds_b128((gptr_v4i)gSrc, (lptr_v4i)ldsDst,
                                                 0, 0);
#else
  *(uint4*)ldsDst = *(const uint4*)gSrc;
#endif
}

__device__ __forceinline__ void wait_async() {
#ifdef HAVE_ASYNC_LDS
#if __has_builtin(__builtin_amdgcn_s_wait_asynccnt)
  __builtin_amdgcn_s_wait_asynccnt(0);
#else
  asm volatile("s_wait_asynccnt 0" ::: "memory");
#endif
#endif
}

// ---------------------------------------------------------------------------
// Tiny prep kernels
// ---------------------------------------------------------------------------
__global__ void silu_k(const float* __restrict__ c, float* __restrict__ out, int n) {
  int i = blockIdx.x * 256 + threadIdx.x;
  if (i < n) { float v = c[i]; out[i] = v / (1.f + __expf(-v)); }
}

// mod = silu(c) @ w_ada + b_ada   (2 x 4608 outputs, K=768)
__global__ void ada_k(const float* __restrict__ sc, const float* __restrict__ w,
                      const float* __restrict__ bias, float* __restrict__ mod) {
  int j = blockIdx.x * 256 + threadIdx.x;           // 0..9215
  int b = j / (6 * CH), col = j % (6 * CH);
  float acc = 0.f;
  for (int k = 0; k < CH; ++k)
    acc += sc[b * CH + k] * w[(size_t)k * (6 * CH) + col];
  mod[j] = acc + bias[col];
}

// f32 [R][Cc] -> f16 [Cc][R]  (weights become row-major in K)
__global__ void transpose_f16_k(const float* __restrict__ in,
                                _Float16* __restrict__ out, int R, int Cc) {
  __shared__ float tile[32][33];
  int bx = blockIdx.x * 32, by = blockIdx.y * 32;
  int tx = threadIdx.x, ty = threadIdx.y;
  for (int i = ty; i < 32; i += 8)
    tile[i][tx] = in[(size_t)(by + i) * Cc + bx + tx];
  __syncthreads();
  for (int i = ty; i < 32; i += 8)
    out[(size_t)(bx + i) * R + by + tx] = (_Float16)tile[tx][i];
}

// rmsnorm + modulate -> f16 GEMM operand.  One block per token row.
__global__ __launch_bounds__(256) void modulate_k(
    const float* __restrict__ xin, const float* __restrict__ g,
    const float* __restrict__ mod, int shOff, int scOff,
    _Float16* __restrict__ hout) {
  int s = blockIdx.x;
  const float* xr = xin + (size_t)s * CH;
  float part = 0.f;
  for (int j = threadIdx.x; j < CH; j += 256) { float v = xr[j]; part += v * v; }
  __shared__ float red[8];
  for (int m = 16; m >= 1; m >>= 1) part += __shfl_xor(part, m, 32);
  if ((threadIdx.x & 31) == 0) red[threadIdx.x >> 5] = part;
  __syncthreads();
  float tot = 0.f;
#pragma unroll
  for (int i = 0; i < 8; ++i) tot += red[i];
  float rs = rsqrtf(tot / CH + EPSF);
  int b = s / SEQ;
  const float* mrow = mod + (size_t)b * 6 * CH;
  for (int j = threadIdx.x; j < CH; j += 256) {
    float t = xr[j] * rs * g[j];
    hout[(size_t)s * CH + j] = (_Float16)(t * (1.f + mrow[scOff + j]) + mrow[shOff + j]);
  }
}

// per-(b,h,n): head rmsnorm on q/k, RoPE.  Emits q/k f16 [B,H,N,D] and V
// f16 *pre-transposed* [B,H,D,N] so the attention kernel can async-copy both
// operand tiles contiguously.  Lane i owns rotation pair (2i, 2i+1).
__global__ __launch_bounds__(256) void qkv_post_k(
    const float* __restrict__ qkv, const float* __restrict__ g_qn,
    const float* __restrict__ g_kn, _Float16* __restrict__ q16,
    _Float16* __restrict__ k16, _Float16* __restrict__ vT16) {
  int w = blockIdx.x * 8 + (threadIdx.x >> 5);
  int lane = threadIdx.x & 31;
  int n = w % SEQ; int bh = w / SEQ; int b = bh / NH; int h = bh % NH;
  size_t srow = (size_t)(b * SEQ + n) * (3 * CH) + h * HD;
  int d0 = lane * 2, d1 = d0 + 1;
  float ang = (float)n * __powf(10000.f, -(float)d0 / (float)HD);
  float cs = __cosf(ang), sn = __sinf(ang);
  size_t obase = (size_t)bh * SEQ * HD + (size_t)n * HD;
  { // Q
    float a0 = qkv[srow + d0], a1 = qkv[srow + d1];
    float ss = a0 * a0 + a1 * a1;
    for (int m = 16; m >= 1; m >>= 1) ss += __shfl_xor(ss, m, 32);
    float rs = rsqrtf(ss / HD + EPSF);
    a0 *= rs * g_qn[d0]; a1 *= rs * g_qn[d1];
    q16[obase + d0] = (_Float16)(a0 * cs - a1 * sn);
    q16[obase + d1] = (_Float16)(a0 * sn + a1 * cs);
  }
  { // K
    float a0 = qkv[srow + CH + d0], a1 = qkv[srow + CH + d1];
    float ss = a0 * a0 + a1 * a1;
    for (int m = 16; m >= 1; m >>= 1) ss += __shfl_xor(ss, m, 32);
    float rs = rsqrtf(ss / HD + EPSF);
    a0 *= rs * g_kn[d0]; a1 *= rs * g_kn[d1];
    k16[obase + d0] = (_Float16)(a0 * cs - a1 * sn);
    k16[obase + d1] = (_Float16)(a0 * sn + a1 * cs);
  }
  // V transposed: vT[bh][d][n]
  size_t vtb = (size_t)bh * HD * SEQ;
  vT16[vtb + (size_t)d0 * SEQ + n] = (_Float16)qkv[srow + 2 * CH + d0];
  vT16[vtb + (size_t)d1 * SEQ + n] = (_Float16)qkv[srow + 2 * CH + d1];
}

// ---------------------------------------------------------------------------
// WMMA GEMM: A f16 [M][K] row-major, Bt f16 [Nout][K] row-major (w transposed).
// 128x128x32 tile, 8 waves (4x2), each wave 32x64 (2x4 16x16 accumulators).
// Double-buffered async global->LDS staging; ONE barrier per K-step.
// MODE 0: out = acc + bias                          -> f32
// MODE 1/3: out = resid + gate * (acc + bias)       -> f32
// MODE 2: SwiGLU fusion: silu(acc1+b1)*(acc2+b2)    -> f16
// ---------------------------------------------------------------------------
template <int MODE>
__global__ __launch_bounds__(256) void gemm_k(
    const _Float16* __restrict__ A, const _Float16* __restrict__ Bt,
    int M, int Nn, int K, const float* __restrict__ bias,
    float* __restrict__ outf, _Float16* __restrict__ outh,
    const float* __restrict__ resid, const float* __restrict__ mod, int gateOff) {
  constexpr int BM = 128, BN = 128, BK = 32, LDP = BK + 8;
  __shared__ _Float16 sA[2][BM][LDP];
  __shared__ _Float16 sB[2][BN][LDP];
  __shared__ _Float16 sB2[(MODE == 2) ? 2 : 1][(MODE == 2) ? BN : 1][LDP];

  const int tid = threadIdx.x;
  const int wave = tid >> 5, lane = tid & 31;
  const int l16 = lane & 15, hl = lane >> 4;
  const int wm = wave >> 1, wn = wave & 1;
  const int bm = blockIdx.y * BM, bn = blockIdx.x * BN;

  const v8f vz = {0.f, 0.f, 0.f, 0.f, 0.f, 0.f, 0.f, 0.f};
  v8f acc[2][4], acc2[2][4];
#pragma unroll
  for (int i = 0; i < 2; ++i)
#pragma unroll
    for (int j = 0; j < 4; ++j) { acc[i][j] = vz; acc2[i][j] = vz; }

  auto issue = [&](int kt, int buf) {
    for (int i = tid; i < BM * BK / 8; i += 256) {
      int r = i >> 2, cc = (i & 3) * 8;
      async_ld16(&sA[buf][r][cc], &A[(size_t)(bm + r) * K + kt + cc]);
    }
    for (int i = tid; i < BN * BK / 8; i += 256) {
      int r = i >> 2, cc = (i & 3) * 8;
      async_ld16(&sB[buf][r][cc], &Bt[(size_t)(bn + r) * K + kt + cc]);
      if constexpr (MODE == 2)
        async_ld16(&sB2[buf][r][cc], &Bt[(size_t)(bn + r + M3) * K + kt + cc]);
    }
  };

  const int nk = K / BK;
  int cur = 0;
  issue(0, 0);
  for (int t = 0; t < nk; ++t) {
    wait_async();
    __syncthreads();                 // tile `cur` ready; prev compute done
    if (t + 1 < nk) issue((t + 1) * BK, cur ^ 1);

    AF af[2], bf[4], bf2[4];
#pragma unroll
    for (int i = 0; i < 2; ++i) {
      // A layout: lane half h reads K = h*8..h*8+7 and 16+h*8..16+h*8+7
      const _Float16* p = &sA[cur][wm * 32 + i * 16 + l16][hl * 8];
      af[i].u[0] = *(const uint4*)p;
      af[i].u[1] = *(const uint4*)(p + 16);
    }
#pragma unroll
    for (int j = 0; j < 4; ++j) {
      // B layout: lane half h reads K = h*8..h*8+7 and 16+h*8..16+h*8+7 for column l16
      const _Float16* p = &sB[cur][wn * 64 + j * 16 + l16][hl * 8];
      bf[j].u[0] = *(const uint4*)p;
      bf[j].u[1] = *(const uint4*)(p + 16);
      if constexpr (MODE == 2) {
        const _Float16* p2 = &sB2[cur][wn * 64 + j * 16 + l16][hl * 8];
        bf2[j].u[0] = *(const uint4*)p2;
        bf2[j].u[1] = *(const uint4*)(p2 + 16);
      }
    }
#pragma unroll
    for (int i = 0; i < 2; ++i)
#pragma unroll
      for (int j = 0; j < 4; ++j) {
        acc[i][j] = wmma16(af[i].v, bf[j].v, acc[i][j]);
        if constexpr (MODE == 2) acc2[i][j] = wmma16(af[i].v, bf2[j].v, acc2[i][j]);
      }
    cur ^= 1;
  }

#pragma unroll
  for (int i = 0; i < 2; ++i)
#pragma unroll
    for (int j = 0; j < 4; ++j)
#pragma unroll
      for (int r = 0; r < 8; ++r) {
        int row = bm + wm * 32 + i * 16 + hl * 8 + r;   // C layout: VGPR r -> M=r / r+8
        int col = bn + wn * 64 + j * 16 + l16;
        float v = acc[i][j][r];
        if constexpr (MODE == 0) {
          outf[(size_t)row * Nn + col] = v + bias[col];
        } else if constexpr (MODE == 1 || MODE == 3) {
          int b = row / SEQ;
          float y = v + bias[col];
          outf[(size_t)row * Nn + col] =
              resid[(size_t)row * Nn + col] + mod[(size_t)b * 6 * CH + gateOff + col] * y;
        } else {  // SwiGLU
          float v1 = v + bias[col];
          float v2 = acc2[i][j][r] + bias[col + M3];
          float hsw = v1 / (1.f + __expf(-v1)) * v2;
          outh[(size_t)row * Nn + col] = (_Float16)hsw;
        }
      }
}

// ---------------------------------------------------------------------------
// Flash attention, one head-slice per block: 4 waves, 64 q rows, kv tiles of
// 64, with double-buffered async K/V^T tile staging.
// ---------------------------------------------------------------------------
__global__ __launch_bounds__(128) void attn_k(
    const _Float16* __restrict__ q16, const _Float16* __restrict__ k16,
    const _Float16* __restrict__ vT16, _Float16* __restrict__ attn16) {
  constexpr int LDP = HD + 8;
  __shared__ _Float16 sK[2][64][LDP];    // K tile row-major [kv][d]
  __shared__ _Float16 sVt[2][64][LDP];   // V^T tile [d][kv]
  __shared__ _Float16 sP[64][LDP];       // probs f16, per-wave 16-row stripes

  const int tid = threadIdx.x;
  const int wave = tid >> 5, lane = tid & 31;
  const int l16 = lane & 15, hl = lane >> 4;
  const int bh = blockIdx.y, h = bh % NH, b = bh / NH;
  const int q0 = blockIdx.x * 64;
  const size_t headBase = (size_t)bh * SEQ * HD;
  const size_t vtBase = (size_t)bh * HD * SEQ;
  const float scale = 0.125f;  // 1/sqrt(64)

  AF qa[2];  // persistent Q A-fragments (two K=32 chunks of D=64)
  {
    const _Float16* qrow = q16 + headBase + (size_t)(q0 + wave * 16 + l16) * HD;
#pragma unroll
    for (int c = 0; c < 2; ++c) {
      qa[c].u[0] = *(const uint4*)(qrow + c * 32 + hl * 8);
      qa[c].u[1] = *(const uint4*)(qrow + c * 32 + hl * 8 + 16);
    }
  }

  const v8f vz = {0.f, 0.f, 0.f, 0.f, 0.f, 0.f, 0.f, 0.f};
  float mrow[8], lrow[8];
  v8f o[4];
#pragma unroll
  for (int r = 0; r < 8; ++r) { mrow[r] = -1e30f; lrow[r] = 0.f; }
#pragma unroll
  for (int t = 0; t < 4; ++t) o[t] = vz;

  auto issue = [&](int kv0, int buf) {
    for (int i = tid; i < 64 * 8; i += 128) {
      int r = i >> 3, cc = (i & 7) * 8;
      async_ld16(&sK[buf][r][cc], &k16[headBase + (size_t)(kv0 + r) * HD + cc]);
      async_ld16(&sVt[buf][r][cc], &vT16[vtBase + (size_t)r * SEQ + kv0 + cc]);
    }
  };

  int cur = 0;
  issue(0, 0);
  for (int kb = 0; kb < SEQ / 64; ++kb) {
    wait_async();
    __syncthreads();                 // tile `cur` ready; prev PV reads done
    if (kb + 1 < SEQ / 64) issue((kb + 1) * 64, cur ^ 1);

    v8f s[4];
#pragma unroll
    for (int nt = 0; nt < 4; ++nt) {
      s[nt] = vz;
#pragma unroll
      for (int c = 0; c < 2; ++c) {
        AF bfr;
        const _Float16* p = &sK[cur][nt * 16 + l16][c * 32 + hl * 8];
        bfr.u[0] = *(const uint4*)p; bfr.u[1] = *(const uint4*)(p + 16);
        s[nt] = wmma16(qa[c].v, bfr.v, s[nt]);
      }
    }

    // online softmax (lane halves hold rows hl*8+r, cols across 16 lanes)
    float mnew[8];
#pragma unroll
    for (int r = 0; r < 8; ++r) mnew[r] = mrow[r];
#pragma unroll
    for (int nt = 0; nt < 4; ++nt)
#pragma unroll
      for (int r = 0; r < 8; ++r) mnew[r] = fmaxf(mnew[r], s[nt][r] * scale);
#pragma unroll
    for (int r = 0; r < 8; ++r)
      for (int msk = 1; msk < 16; msk <<= 1)
        mnew[r] = fmaxf(mnew[r], __shfl_xor(mnew[r], msk, 32));

    float corr[8], psum[8];
#pragma unroll
    for (int r = 0; r < 8; ++r) {
      corr[r] = __expf(mrow[r] - mnew[r]); psum[r] = 0.f; mrow[r] = mnew[r];
    }
#pragma unroll
    for (int nt = 0; nt < 4; ++nt)
#pragma unroll
      for (int r = 0; r < 8; ++r) {
        float p = __expf(s[nt][r] * scale - mnew[r]);
        psum[r] += p;
        sP[wave * 16 + hl * 8 + r][nt * 16 + l16] = (_Float16)p;
      }
#pragma unroll
    for (int r = 0; r < 8; ++r) {
      for (int msk = 1; msk < 16; msk <<= 1) psum[r] += __shfl_xor(psum[r], msk, 32);
      lrow[r] = lrow[r] * corr[r] + psum[r];
    }
#pragma unroll
    for (int t = 0; t < 4; ++t)
#pragma unroll
      for (int r = 0; r < 8; ++r) o[t][r] *= corr[r];

    __syncthreads();                 // sP stripe visible across lanes

    // O += P @ V
#pragma unroll
    for (int c = 0; c < 2; ++c) {
      AF pa;
      const _Float16* pp = &sP[wave * 16 + l16][c * 32 + hl * 8];
      pa.u[0] = *(const uint4*)pp; pa.u[1] = *(const uint4*)(pp + 16);
#pragma unroll
      for (int t = 0; t < 4; ++t) {
        AF bv;
        const _Float16* vp = &sVt[cur][t * 16 + l16][c * 32 + hl * 8];
        bv.u[0] = *(const uint4*)vp; bv.u[1] = *(const uint4*)(vp + 16);
        o[t] = wmma16(pa.v, bv.v, o[t]);
      }
    }
    cur ^= 1;
  }

#pragma unroll
  for (int t = 0; t < 4; ++t)
#pragma unroll
    for (int r = 0; r < 8; ++r) {
      int qrow = q0 + wave * 16 + hl * 8 + r;
      int col = h * HD + t * 16 + l16;
      attn16[((size_t)b * SEQ + qrow) * CH + col] = (_Float16)(o[t][r] / lrow[r]);
    }
}

// ---------------------------------------------------------------------------
extern "C" void kernel_launch(void* const* d_in, const int* in_sizes, int n_in,
                              void* d_out, int out_size, void* d_ws, size_t ws_size,
                              hipStream_t stream) {
  const float* x      = (const float*)d_in[0];
  const float* cvec   = (const float*)d_in[1];
  const float* g1     = (const float*)d_in[2];
  const float* g2     = (const float*)d_in[3];
  const float* w_ada  = (const float*)d_in[4];
  const float* b_ada  = (const float*)d_in[5];
  const float* w_qkv  = (const float*)d_in[6];
  const float* b_qkv  = (const float*)d_in[7];
  const float* g_qn   = (const float*)d_in[8];
  const float* g_kn   = (const float*)d_in[9];
  const float* w_proj = (const float*)d_in[10];
  const float* b_proj = (const float*)d_in[11];
  const float* w12    = (const float*)d_in[12];
  const float* b12    = (const float*)d_in[13];
  const float* w3     = (const float*)d_in[14];
  const float* b3     = (const float*)d_in[15];
  float* xout = (float*)d_out;

  char* ws = (char*)d_ws;
  size_t off = 0;
  auto alloc = [&](size_t bytes) { size_t o = off; off = (off + bytes + 255) & ~255ULL; return o; };
  float*     siluc   = (float*)    (ws + alloc((size_t)BD * CH * 4));
  float*     modp    = (float*)    (ws + alloc((size_t)BD * 6 * CH * 4));
  _Float16*  h16     = (_Float16*) (ws + alloc((size_t)SROWS * CH * 2));
  _Float16*  wqkvT   = (_Float16*) (ws + alloc((size_t)(3 * CH) * CH * 2));
  _Float16*  wprojT  = (_Float16*) (ws + alloc((size_t)CH * CH * 2));
  _Float16*  w12T    = (_Float16*) (ws + alloc((size_t)(2 * M3) * CH * 2));
  _Float16*  w3T     = (_Float16*) (ws + alloc((size_t)CH * M3 * 2));
  float*     qkvf    = (float*)    (ws + alloc((size_t)SROWS * 3 * CH * 4));
  _Float16*  q16     = (_Float16*) (ws + alloc((size_t)BD * NH * SEQ * HD * 2));
  _Float16*  k16     = (_Float16*) (ws + alloc((size_t)BD * NH * SEQ * HD * 2));
  _Float16*  vT16    = (_Float16*) (ws + alloc((size_t)BD * NH * HD * SEQ * 2));
  _Float16*  attn16  = (_Float16*) (ws + alloc((size_t)SROWS * CH * 2));
  float*     x1      = (float*)    (ws + alloc((size_t)SROWS * CH * 4));
  _Float16*  hid16   = (_Float16*) (ws + alloc((size_t)SROWS * M3 * 2));
  (void)ws_size; (void)in_sizes; (void)n_in; (void)out_size;

  // 1) adaLN modulation vector
  silu_k<<<(BD * CH + 255) / 256, 256, 0, stream>>>(cvec, siluc, BD * CH);
  ada_k<<<(BD * 6 * CH) / 256, 256, 0, stream>>>(siluc, w_ada, b_ada, modp);

  // 2) weight convert+transpose to f16 [Nout][K]
  dim3 tb(32, 8);
  transpose_f16_k<<<dim3((3 * CH) / 32, CH / 32), tb, 0, stream>>>(w_qkv, wqkvT, CH, 3 * CH);
  transpose_f16_k<<<dim3(CH / 32, CH / 32), tb, 0, stream>>>(w_proj, wprojT, CH, CH);
  transpose_f16_k<<<dim3((2 * M3) / 32, CH / 32), tb, 0, stream>>>(w12, w12T, CH, 2 * M3);
  transpose_f16_k<<<dim3(CH / 32, M3 / 32), tb, 0, stream>>>(w3, w3T, M3, CH);

  // 3) h = modulate(rmsnorm(x,g1), sh_msa, sc_msa) -> f16
  modulate_k<<<SROWS, 256, 0, stream>>>(x, g1, modp, 0, CH, h16);

  // 4) qkv GEMM (4096x2304x768)
  gemm_k<0><<<dim3((3 * CH) / 128, SROWS / 128), 256, 0, stream>>>(
      h16, wqkvT, SROWS, 3 * CH, CH, b_qkv, qkvf, nullptr, nullptr, nullptr, 0);

  // 5) q/k rmsnorm + rope; q/k -> [B,H,N,D], V -> [B,H,D,N] (pre-transposed)
  qkv_post_k<<<(BD * NH * SEQ) / 8, 256, 0, stream>>>(qkvf, g_qn, g_kn, q16, k16, vT16);

  // 6) flash attention
  attn_k<<<dim3(SEQ / 64, BD * NH), 128, 0, stream>>>(q16, k16, vT16, attn16);

  // 7) x1 = x + g_msa * (attn @ w_proj + b_proj)
  gemm_k<1><<<dim3(CH / 128, SROWS / 128), 256, 0, stream>>>(
      attn16, wprojT, SROWS, CH, CH, b_proj, x1, nullptr, x, modp, 2 * CH);

  // 8) h2 = modulate(rmsnorm(x1,g2), sh_mlp, sc_mlp) -> f16 (reuse h16)
  modulate_k<<<SROWS, 256, 0, stream>>>(x1, g2, modp, 3 * CH, 4 * CH, h16);

  // 9) hidden = silu(h2@w1+b1) * (h2@w2+b2)  (fused SwiGLU GEMM) -> f16
  gemm_k<2><<<dim3(M3 / 128, SROWS / 128), 256, 0, stream>>>(
      h16, w12T, SROWS, M3, CH, b12, nullptr, hid16, nullptr, nullptr, 0);

  // 10) out = x1 + g_mlp * (hidden @ w3 + b3)
  gemm_k<3><<<dim3(CH / 128, SROWS / 128), 256, 0, stream>>>(
      hid16, w3T, SROWS, CH, M3, b3, xout, nullptr, x1, modp, 5 * CH);
}
